// SupConLoss_48696339202756
// MI455X (gfx1250) — matrix operand order repaired and checked by reading, not compiled, not run on hardware
//
#include <hip/hip_runtime.h>
#include <hip/hip_fp16.h>

// ---------------------------------------------------------------------------
// SupCon-style loss, fused for MI455X (gfx1250, wave32, WMMA f16->f32 K=32).
//
//   0) convert f32 features/pools -> f16; bit-pack+transpose neglect mask
//   1) R2[j] = {max_k f_j.pool_ag_k, max_k s_j.pool_sp_k}  via WMMA + per-lane
//      running max (one cross-lane reduction at end)
//   2) fused 4x Gram (f, s_pr, z, z_mix): 16-row strips x 4096 cols.
//      Mask logic + UNSHIFTED sum-exp (max-shift cancels analytically and
//      |L|<=~14.3 so f32 exp is safe) accumulated PER LANE; hot loop is
//      pure b128 loads + v_wmma + thin VALU tail (1 u16 + 1 b64 side load).
//   3) single-block finalize -> scalar loss
// ---------------------------------------------------------------------------

typedef __attribute__((ext_vector_type(16))) _Float16 v16h;
typedef __attribute__((ext_vector_type(8)))  _Float16 v8h;
typedef __attribute__((ext_vector_type(4)))  _Float16 v4h;
typedef __attribute__((ext_vector_type(8)))  float    v8f;

#define BATCH   4096
#define DDIM    128
#define POOLN   1024
#define SPLIT   256
#define MBLK    16           // BATCH / SPLIT
#define TEMP_F  0.07f
#define LMBD_F  0.5f

// ---------------- fragment loaders (row-major f16, row stride DDIM) --------
// A (16x32, f16): lane L holds row (L&15); lanes<16 get K klo..klo+15,
// lanes>=16 get klo+16..klo+31 (klo = kk*32).
__device__ __forceinline__ v16h load_a_frag(const _Float16* __restrict__ X,
                                            int rowbase, int l16, int h, int kk) {
  union { v16h v; v8h p[2]; } u;
  const _Float16* p = X + (size_t)(rowbase + l16) * DDIM + kk * 32 + h * 16;
  u.p[0] = *(const v8h*)(p);        // K klo+16h .. +7   -> elems 0..7
  u.p[1] = *(const v8h*)(p + 8);    // K klo+16h+8..+15 -> elems 8..15
  return u.v;
}
// B (32x16, f16): lane L holds column (L&15); lanes<16 get K kk*32..+15,
// lanes>=16 get K kk*32+16..+31 -> 16 contiguous halves (32B aligned).
__device__ __forceinline__ v16h load_b_frag(const _Float16* __restrict__ X,
                                            int colbase, int l16, int h, int kk) {
  return *(const v16h*)(X + (size_t)(colbase + l16) * DDIM + kk * 32 + h * 16);
}

// 16-lane-group butterfly reductions (used only once per kernel, at the end)
__device__ __forceinline__ float red16_add(float x) {
  #pragma unroll
  for (int o = 1; o < 16; o <<= 1) x += __shfl_xor(x, o, 32);
  return x;
}
__device__ __forceinline__ float red16_max(float x) {
  #pragma unroll
  for (int o = 1; o < 16; o <<= 1) x = fmaxf(x, __shfl_xor(x, o, 32));
  return x;
}

// ---------------- phase 0a: f32 -> f16 -------------------------------------
__global__ void cvt_f32_f16_kernel(const float* __restrict__ src,
                                   _Float16* __restrict__ dst, int n4) {
  int i = blockIdx.x * blockDim.x + threadIdx.x;
  if (i < n4) {
    float4 x = ((const float4*)src)[i];
    v4h o; o.x = (_Float16)x.x; o.y = (_Float16)x.y;
    o.z = (_Float16)x.z; o.w = (_Float16)x.w;
    ((v4h*)dst)[i] = o;
  }
}

// ---------------- phase 0b: transpose + bit-pack neglect mask --------------
// packT[j*16 + g] bit b  =  (maskb[(g*16+b)*BATCH + j] > 0)
__global__ void pack_mask_kernel(const float* __restrict__ mb,
                                 unsigned short* __restrict__ packT) {
  int idx = blockIdx.x * blockDim.x + threadIdx.x;   // over BATCH*16
  if (idx < BATCH * 16) {
    const int j = idx >> 4, g = idx & 15;
    unsigned bits = 0;
    #pragma unroll
    for (int b = 0; b < 16; ++b)
      bits |= (mb[(size_t)(g * 16 + b) * BATCH + j] > 0.f) ? (1u << b) : 0u;
    packT[idx] = (unsigned short)bits;
  }
}

// ---------------- phase 1: R2[2*r+comp] = max_k feat_r . pool_k ------------
__global__ void pool_rowmax_kernel(const _Float16* __restrict__ feat,
                                   const _Float16* __restrict__ pool,
                                   float* __restrict__ R2, int comp) {
  const int lane = threadIdx.x & 31;
  const int wave = threadIdx.x >> 5;
  const int tile = blockIdx.x * (blockDim.x >> 5) + wave;   // 0..255
  const int rowbase = tile * 16;
  const int h = lane >> 4, l16 = lane & 15;

  v16h a[4];
  #pragma unroll
  for (int kk = 0; kk < 4; ++kk) a[kk] = load_a_frag(feat, rowbase, l16, h, kk);

  float rmax[8];                       // per-lane running max (column-partial)
  #pragma unroll
  for (int v = 0; v < 8; ++v) rmax[v] = -1e30f;

  for (int ct = 0; ct < POOLN / 16; ++ct) {
    v8f acc = {};
    #pragma unroll
    for (int kk = 0; kk < 4; ++kk) {
      v16h b = load_b_frag(pool, ct * 16, l16, h, kk);
      acc = __builtin_amdgcn_wmma_f32_16x16x32_f16(false, a[kk], false, b,
                                                   (short)0, acc, false, false);
    }
    #pragma unroll
    for (int v = 0; v < 8; ++v) rmax[v] = fmaxf(rmax[v], acc[v]);
  }
  #pragma unroll
  for (int v = 0; v < 8; ++v) {
    float m = red16_max(rmax[v]);
    if (l16 == 0) R2[2 * (rowbase + v + 8 * h) + comp] = m;
  }
}

// ---------------- phase 2: fused Gram + masks + sum-exp --------------------
// stats[r] = { wsum, valid, mlpp_z, mlpp_zmix }
__global__ void fused_gram_kernel(const _Float16* __restrict__ fh,
                                  const _Float16* __restrict__ sh,
                                  const _Float16* __restrict__ zh,
                                  const _Float16* __restrict__ zmh,
                                  const float* __restrict__ R2,     // [BATCH][2]
                                  const unsigned short* __restrict__ packT, // [BATCH][16]
                                  float4* __restrict__ stats) {
  const int lane = threadIdx.x & 31;
  const int wave = threadIdx.x >> 5;
  const int tile = blockIdx.x * (blockDim.x >> 5) + wave;   // 0..255
  const int rowbase = tile * 16;
  const int h = lane >> 4, l16 = lane & 15;
  const int g = tile & 15;             // 16-row group inside SPLIT

  float Rr_ag[8], Rr_sp[8];
  // per-lane column-partial accumulators (reduced across lanes once, at end)
  float wsum[8], c1[8], wLz[8], wLm[8], Zz[8], Zm[8];
  #pragma unroll
  for (int v = 0; v < 8; ++v) {
    const int r = rowbase + v + 8 * h;
    Rr_ag[v] = R2[2 * r]; Rr_sp[v] = R2[2 * r + 1];
    wsum[v] = 0.f; c1[v] = 0.f; wLz[v] = 0.f; wLm[v] = 0.f;
    Zz[v] = 0.f; Zm[v] = 0.f;
  }

  for (int ct = 0; ct < BATCH / 16; ++ct) {
    const int j = ct * 16 + l16;               // this lane's column
    const float2 Rj = ((const float2*)R2)[j];  // one b64 load
    const unsigned mb8 = (unsigned)packT[j * 16 + g] >> (8 * h); // one u16 load

    v8f cF = {}, cS = {}, cZ = {}, cM = {};
    #pragma unroll
    for (int kk = 0; kk < 4; ++kk) {
      v16h aF = load_a_frag(fh,  rowbase, l16, h, kk);
      v16h aS = load_a_frag(sh,  rowbase, l16, h, kk);
      v16h aZ = load_a_frag(zh,  rowbase, l16, h, kk);
      v16h aM = load_a_frag(zmh, rowbase, l16, h, kk);
      v16h bF = load_b_frag(fh,  ct * 16, l16, h, kk);
      v16h bS = load_b_frag(sh,  ct * 16, l16, h, kk);
      v16h bZ = load_b_frag(zh,  ct * 16, l16, h, kk);
      v16h bM = load_b_frag(zmh, ct * 16, l16, h, kk);
      cF = __builtin_amdgcn_wmma_f32_16x16x32_f16(false, aF, false, bF, (short)0, cF, false, false);
      cS = __builtin_amdgcn_wmma_f32_16x16x32_f16(false, aS, false, bS, (short)0, cS, false, false);
      cZ = __builtin_amdgcn_wmma_f32_16x16x32_f16(false, aZ, false, bZ, (short)0, cZ, false, false);
      cM = __builtin_amdgcn_wmma_f32_16x16x32_f16(false, aM, false, bM, (short)0, cM, false, false);
    }

    #pragma unroll
    for (int v = 0; v < 8; ++v) {
      const int r = rowbase + v + 8 * h;
      const float cf = cF[v], cs = cS[v];
      const bool m1b = (cf > Rr_ag[v]) || (cf > Rj.x);
      const bool meb = (cs > Rr_sp[v]) || (cs > Rj.y);
      const bool neg = (j != r);
      const bool nb  = (mb8 >> v) & 1u;
      const float negl = ((m1b || nb) && neg) ? 1.f : 0.f;
      const float m1 = (m1b && neg) ? 1.f : 0.f;
      const float me = (meb && neg) ? 1.f : 0.f;
      const float w  = m1 + LMBD_F * me;
      const float Lz = cZ[v] * (1.0f / TEMP_F);   // |L| <= ~14.3: exp safe
      const float Lm = cM[v] * (1.0f / TEMP_F);

      wsum[v] += w;
      c1[v]   += m1;
      wLz[v]  += w * Lz;
      wLm[v]  += w * Lm;
      Zz[v]   += negl * __expf(Lz);
      Zm[v]   += negl * __expf(Lm);
    }
  }

  // single cross-lane reduction pass per row
  #pragma unroll
  for (int v = 0; v < 8; ++v) {
    const float ws  = red16_add(wsum[v]);
    const float cc  = red16_add(c1[v]);
    const float wlz = red16_add(wLz[v]);
    const float wlm = red16_add(wLm[v]);
    const float zz  = red16_add(Zz[v]);
    const float zm  = red16_add(Zm[v]);
    if (l16 == 0) {
      const int r = rowbase + v + 8 * h;
      const float d  = (ws > 0.f) ? ws : 1.f;
      const float lz = (zz > 0.f) ? __logf(zz) : 0.f;
      const float lm = (zm > 0.f) ? __logf(zm) : 0.f;
      stats[r] = make_float4(ws, (cc > 0.f) ? 1.f : 0.f,
                             (wlz - lz * ws) / d,
                             (wlm - lm * ws) / d);
    }
  }
}

// ---------------- phase 3: per-block reweighting -> scalar -----------------
__device__ __forceinline__ float block_red_add(float v, float* red) {
  const int t = threadIdx.x;
  red[t] = v; __syncthreads();
  for (int s = SPLIT / 2; s > 0; s >>= 1) {
    if (t < s) red[t] += red[t + s];
    __syncthreads();
  }
  float r = red[0]; __syncthreads();
  return r;
}

__global__ void finalize_kernel(const float4* __restrict__ stats,
                                float* __restrict__ out) {
  __shared__ float red[SPLIT];
  const int t = threadIdx.x;
  float loss = 0.f;
  for (int m = 0; m < MBLK; ++m) {
    const float4 st = stats[m * SPLIT + t];
    const float ws = st.x, val = st.y;
    const float s  = (val > 0.f) ? ws : 0.f;
    const float ssum = block_red_add(s, red);
    float nv = block_red_add(val, red);
    nv = fmaxf(nv, 1.f);
    const float pnm = (ssum > 0.f) ? (s / ssum * nv) : 0.f;
    const float bz = block_red_add(st.z * pnm, red) / nv;
    const float bm = block_red_add(st.w * pnm, red) / nv;
    loss += bz + bm;
  }
  if (t == 0) out[0] = -loss / (float)MBLK / 2.0f;
}

// ---------------------------------------------------------------------------
extern "C" void kernel_launch(void* const* d_in, const int* in_sizes, int n_in,
                              void* d_out, int out_size, void* d_ws, size_t ws_size,
                              hipStream_t stream) {
  (void)in_sizes; (void)n_in; (void)out_size; (void)ws_size;
  const float* z   = (const float*)d_in[0];
  const float* spr = (const float*)d_in[1];
  const float* f   = (const float*)d_in[2];
  const float* pa  = (const float*)d_in[3];
  const float* ps  = (const float*)d_in[4];
  const float* zm  = (const float*)d_in[5];
  const float* mb  = (const float*)d_in[6];

  const size_t FEAT = (size_t)BATCH * DDIM;   // 524288
  const size_t PO   = (size_t)POOLN * DDIM;   // 131072

  char* w = (char*)d_ws;
  _Float16* fh  = (_Float16*)w; w += FEAT * 2;
  _Float16* sh  = (_Float16*)w; w += FEAT * 2;
  _Float16* zh  = (_Float16*)w; w += FEAT * 2;
  _Float16* zmh = (_Float16*)w; w += FEAT * 2;
  _Float16* pah = (_Float16*)w; w += PO * 2;
  _Float16* psh = (_Float16*)w; w += PO * 2;
  float* R2 = (float*)w;                     w += (size_t)BATCH * 2 * 4;
  unsigned short* packT = (unsigned short*)w; w += (size_t)BATCH * 16 * 2;
  float4* stats = (float4*)w;                 // BATCH * 16 bytes

  // phase 0: convert (vectorized x4) + mask pack
  const int T = 256;
  cvt_f32_f16_kernel<<<(int)(FEAT / 4 + T - 1) / T, T, 0, stream>>>(f,   fh,  (int)(FEAT / 4));
  cvt_f32_f16_kernel<<<(int)(FEAT / 4 + T - 1) / T, T, 0, stream>>>(spr, sh,  (int)(FEAT / 4));
  cvt_f32_f16_kernel<<<(int)(FEAT / 4 + T - 1) / T, T, 0, stream>>>(z,   zh,  (int)(FEAT / 4));
  cvt_f32_f16_kernel<<<(int)(FEAT / 4 + T - 1) / T, T, 0, stream>>>(zm,  zmh, (int)(FEAT / 4));
  cvt_f32_f16_kernel<<<(int)(PO / 4 + T - 1) / T,   T, 0, stream>>>(pa,  pah, (int)(PO / 4));
  cvt_f32_f16_kernel<<<(int)(PO / 4 + T - 1) / T,   T, 0, stream>>>(ps,  psh, (int)(PO / 4));
  pack_mask_kernel<<<(BATCH * 16 + T - 1) / T, T, 0, stream>>>(mb, packT);

  // phase 1: pool row-max (256 row tiles, 4 waves/block)
  pool_rowmax_kernel<<<64, 128, 0, stream>>>(fh, pah, R2, 0);
  pool_rowmax_kernel<<<64, 128, 0, stream>>>(sh, psh, R2, 1);

  // phase 2: fused 4x Gram + masks + sum-exp
  fused_gram_kernel<<<64, 128, 0, stream>>>(fh, sh, zh, zmh, R2, packT, stats);

  // phase 3: scalar loss
  finalize_kernel<<<1, SPLIT, 0, stream>>>(stats, (float*)d_out);
}
